// FEDformer_17549236371550
// MI455X (gfx1250) — matrix operand order repaired and checked by reading, not compiled, not run on hardware
//
#include <hip/hip_runtime.h>
#include <hip/hip_bf16.h>
#include <math.h>

// ---------------------------------------------------------------------------
// FEDformer frequency-domain block for MI455X (gfx1250, wave32, WMMA + TDM).
//   1) cvt x -> bf16; cvt+transpose Wq, Wo -> bf16 [n][k]
//   2) GEMM (bf16 WMMA, f32 acc, TDM weight-tile load): q = x@Wq + bq
//      scattered to [b,h,i,s,ws]
//   3) pruned rDFT over ws (64 -> 16 complex modes)   [f32 WMMA 16x16x4]
//   4) pruned 2D DFT over (t,hs) at 64 panel freqs    [VALU, sincos-bound]
//   5) complex mode mixing with w                     [VALU]
//   6) hermitian fold (VALU) + inverse cos transform  [f32 WMMA 16x16x4]
//   7) GEMM (bf16 WMMA, f32 acc, TDM): out = o@Wo + bo
// ---------------------------------------------------------------------------

typedef __attribute__((ext_vector_type(16))) __bf16       v16bf;
typedef __attribute__((ext_vector_type(8)))  float        v8f;
typedef __attribute__((ext_vector_type(2)))  float        v2f;
typedef __attribute__((ext_vector_type(4)))  unsigned int u32x4;
typedef __attribute__((ext_vector_type(8)))  int          i32x8;
typedef __attribute__((ext_vector_type(4)))  int          i32x4;

#define PI_F 3.14159265358979323846f

// ---------------- fp32 -> bf16 convert ----------------
__global__ void cvt_f32_bf16(const float* __restrict__ in, __bf16* __restrict__ out, int n)
{
    int i = blockIdx.x * 256 + threadIdx.x;
    if (i < n) out[i] = (__bf16)in[i];
}

// ---------------- fp32 (k,n) -> bf16 transposed (n,k), 256x256 ----------------
__global__ void cvt_transpose_bf16(const float* __restrict__ in, __bf16* __restrict__ out)
{
    int i = blockIdx.x * 256 + threadIdx.x;   // 65536 elements
    int k = i >> 8, n = i & 255;
    out[(size_t)n * 256 + k] = (__bf16)in[i];
}

// ---------------- bf16 WMMA GEMM: C[M x 256] = A[M x 256] @ W[256 x 256] + bias ---
// Wt is pre-transposed [n][k] bf16 so the per-block 64x256 tile is a clean 2D
// TDM descriptor (tile_dim0 = 256 contiguous k, tile_dim1 = 64 rows).
// mode 0: C[m*256+n]            (final projection -> d_out)
// mode 1: scatter to Q2[(b*8+h)*32+i][s][ws]  (q projection, pre-transposed)
__launch_bounds__(128)
__global__ void gemm256_wmma(const __bf16* __restrict__ A,
                             const __bf16* __restrict__ Wt,  // [n][k] bf16
                             const float*  __restrict__ bias,
                             float*        __restrict__ C,
                             int mode)
{
    __shared__ __bf16 lds_wt[64 * 256];            // W^T tile: [n_local][k], 32 KB
    const int tid  = threadIdx.x;
    const int lane = tid & 31;
    const int wave = tid >> 5;
    const int n0   = blockIdx.y * 64;
    const int m0   = blockIdx.x * 64 + wave * 16;

    // --- Tensor Data Mover: async-load the 64x256 bf16 weight tile into LDS ---
    if (wave == 0) {
        unsigned long long ga  = (unsigned long long)(const void*)(Wt + (size_t)n0 * 256);
        unsigned int       ldo = (unsigned int)(unsigned long long)(void*)&lds_wt[0];
        u32x4 g0;
        g0[0] = 1u;                                          // count=1, user D#
        g0[1] = ldo;                                         // lds_addr
        g0[2] = (unsigned int)(ga & 0xffffffffu);            // global_addr[31:0]
        g0[3] = (unsigned int)((ga >> 32) & 0x01ffffffu)     // global_addr[56:32]
              | (2u << 30);                                  // type = 2 (image)
        i32x8 g1;
        g1[0] = (int)(1u << 16);      // workgroup_mask=0, data_size=1 (2 bytes)
        g1[1] = (int)(256u << 16);    // tensor_dim0[15:0]=256 in bits 63:48
        g1[2] = (int)(256u << 16);    // tensor_dim1[15:0]=256 in bits 95:80
        g1[3] = (int)(256u << 16);    // tile_dim0=256 in bits 127:112
        g1[4] = 64;                   // tile_dim1=64, tile_dim2=0
        g1[5] = 256;                  // tensor_dim0_stride=256 (elements)
        g1[6] = 0;
        g1[7] = 0;
        i32x4 gz4 = {0, 0, 0, 0};
        i32x8 gz8 = {0, 0, 0, 0, 0, 0, 0, 0};
        __builtin_amdgcn_tensor_load_to_lds(g0, g1, gz4, gz4, gz8, 0);
        __builtin_amdgcn_s_wait_tensorcnt(0);
    }
    __syncthreads();

    v8f acc[4] = {};

    const int     row   = m0 + (lane & 15);
    const int     half8 = (lane >> 4) * 8;        // A-layout: lanes>=16 hold K+8 block
    const __bf16* arow  = A + (size_t)row * 256;

    union Frag { u32x4 u[2]; v16bf v; };

    for (int kt = 0; kt < 8; ++kt) {
        const int k0 = kt * 32;
        Frag af;
        af.u[0] = *(const u32x4*)(arow + k0 + half8);        // K = k0+half8 .. +7
        af.u[1] = *(const u32x4*)(arow + k0 + 16 + half8);   // K = k0+16+half8 .. +7
        if (kt < 7) __builtin_prefetch(arow + k0 + 32, 0, 1);
        #pragma unroll
        for (int nt = 0; nt < 4; ++nt) {
            const __bf16* brow =
                lds_wt + (nt * 16 + (lane & 15)) * 256 + k0 + (lane >> 4) * 8;
            Frag bfr;
            bfr.u[0] = *(const u32x4*)(brow);
            bfr.u[1] = *(const u32x4*)(brow + 16);
            acc[nt] = __builtin_amdgcn_wmma_f32_16x16x32_bf16(
                false, af.v, false, bfr.v, (short)0, acc[nt], false, false);
        }
    }

    // epilogue: C layout — element v: row m0+v+8*(lane>=16), col n0+nt*16+lane%16
    #pragma unroll
    for (int nt = 0; nt < 4; ++nt) {
        #pragma unroll
        for (int v = 0; v < 8; ++v) {
            int   m   = m0 + v + 8 * (lane >> 4);
            int   n   = n0 + nt * 16 + (lane & 15);
            float val = acc[nt][v] + bias[n];
            if (mode == 0) {
                C[(size_t)m * 256 + n] = val;
            } else {
                int b  = m >> 15;            // m = ((b*16+t)*32+hs)*64+ws
                int s  = (m >> 6) & 511;     // s = t*32+hs
                int ws = m & 63;
                int h  = n >> 5;
                int ii = n & 31;
                C[(((size_t)((b * 8 + h) * 32 + ii) * 512 + s) * 64) + ws] = val;
            }
        }
    }
}

// ---------------- stage 1: pruned rDFT over ws via f32 WMMA ----------------
// Per (b,h,i): G[512x32] = Q[512x64] @ [cos|sin][64x32]  (K-loop of 16 x K=4)
// Q2: [1024][512][64] f32,  G: [1024][512][16][2] f32
__launch_bounds__(128)
__global__ void dft_ws_wmma(const float* __restrict__ Q2, float* __restrict__ G)
{
    __shared__ float w2t[32 * 64];  // [n][k]; n<16: cos(f=n), n>=16: sin(f=n-16)
    const int t = threadIdx.x, lane = t & 31, wave = t >> 5;
    for (int idx = t; idx < 2048; idx += 128) {
        int n = idx >> 6, k = idx & 63;
        int f = n & 15;
        float ang = -2.0f * PI_F * (float)(k * f) / 64.0f;
        float sn, c; __sincosf(ang, &sn, &c);
        w2t[idx] = (n & 16) ? sn : c;
    }
    __syncthreads();

    const int bhi = blockIdx.x >> 3;
    const int m0  = (blockIdx.x & 7) * 64 + wave * 16;       // s-row base
    const float* qrow = Q2 + ((size_t)bhi * 512 + m0 + (lane & 15)) * 64;
    const int koffb = (lane >> 4) * 2;   // f32 A/B layout: lane half owns K pair

    v8f accR = {}, accI = {};
    #pragma unroll
    for (int k0 = 0; k0 < 64; k0 += 4) {
        v2f a  = *(const v2f*)(qrow + k0 + koffb);
        v2f br = *(const v2f*)(w2t + (lane & 15) * 64 + k0 + koffb);
        v2f bi = *(const v2f*)(w2t + (16 + (lane & 15)) * 64 + k0 + koffb);
        accR = __builtin_amdgcn_wmma_f32_16x16x4_f32(false, a, false, br,
                                                     (short)0, accR, false, false);
        accI = __builtin_amdgcn_wmma_f32_16x16x4_f32(false, a, false, bi,
                                                     (short)0, accI, false, false);
    }
    #pragma unroll
    for (int v = 0; v < 8; ++v) {
        int s = m0 + v + 8 * (lane >> 4);
        int f = lane & 15;
        size_t o = (((size_t)bhi * 512 + s) * 16 + f) * 2;
        G[o] = accR[v]; G[o + 1] = accI[v];
    }
}

// ---------------- stage 2: pruned 2D DFT over (t,hs) at panel freqs ----------------
// G: [1024][512][16][2] -> Xsel: [1024][64][16][2]
__global__ void dft_panel_kernel(const float* __restrict__ G,
                                 const int*   __restrict__ panel,
                                 float*       __restrict__ Xsel)
{
    extern __shared__ float g[];   // 512*16*2 floats = 64 KB (dynamic)
    const int bhi = blockIdx.x;
    const float* src = G + (size_t)bhi * (512 * 16 * 2);
    for (int i = threadIdx.x; i < 512 * 16 * 2; i += 64) g[i] = src[i];
    __syncthreads();

    const int   p    = threadIdx.x;                 // 64 threads
    const int   flat = panel[p];
    const float fkt  = (float)(flat >> 5) * (2.0f * PI_F / 16.0f);
    const float fkh  = (float)(flat & 31) * (2.0f * PI_F / 32.0f);
    float accr[16], acci[16];
    #pragma unroll
    for (int f = 0; f < 16; ++f) { accr[f] = 0.f; acci[f] = 0.f; }

    for (int s = 0; s < 512; ++s) {
        float ang = -(fkt * (float)(s >> 5) + fkh * (float)(s & 31));
        float sn, c; __sincosf(ang, &sn, &c);
        const float* gr = g + s * 32;
        #pragma unroll
        for (int f = 0; f < 16; ++f) {
            float xr = gr[2 * f], xi = gr[2 * f + 1];
            accr[f] = fmaf(xr, c, fmaf(-xi, sn, accr[f]));
            acci[f] = fmaf(xr, sn, fmaf( xi, c, acci[f]));
        }
    }
    float* out = Xsel + ((size_t)bhi * 64 + p) * 32;
    #pragma unroll
    for (int f = 0; f < 16; ++f) { out[2 * f] = accr[f]; out[2 * f + 1] = acci[f]; }
}

// ---------------- mode mixing: out[b,o] = sum_i x[b,i] * w[h,i,o] per (h,p,f) -----
__global__ void mode_mix_kernel(const float* __restrict__ Xsel,
                                const float* __restrict__ wr,
                                const float* __restrict__ wi,
                                float*       __restrict__ Osel)
{
    __shared__ float xs[4 * 32 * 16 * 2];   // x_sel[b][i][f][2] for this (h,p), 16 KB
    const int h = blockIdx.x >> 6;
    const int p = blockIdx.x & 63;
    const int t = threadIdx.x;
    for (int idx = t; idx < 4096; idx += 256) {
        int c = idx & 1, f = (idx >> 1) & 15, ii = (idx >> 5) & 31, b = idx >> 10;
        xs[idx] = Xsel[(((size_t)((b * 8 + h) * 32 + ii) * 64 + p) * 16 + f) * 2 + c];
    }
    __syncthreads();

    const int f    = t & 15;
    const int slot = t >> 4;
    for (int j = 0; j < 8; ++j) {
        int b = (slot + 16 * j) >> 5, o = (slot + 16 * j) & 31;
        float ar = 0.f, ai = 0.f;
        for (int ii = 0; ii < 32; ++ii) {
            float  xr = xs[((b * 32 + ii) * 16 + f) * 2];
            float  xi = xs[((b * 32 + ii) * 16 + f) * 2 + 1];
            size_t wo = (((size_t)(h * 32 + ii) * 32 + o) * 64 + p) * 16 + f;
            float  a = wr[wo], bb = wi[wo];
            ar = fmaf(xr, a, fmaf(-xi, bb, ar));
            ai = fmaf(xr, bb, fmaf( xi, a, ai));
        }
        size_t oi = (((size_t)((b * 8 + h) * 32 + o) * 64 + p) * 16 + f) * 2;
        Osel[oi] = ar; Osel[oi + 1] = ai;
    }
}

// ---------------- inverse: hermitian fold (VALU) + cos transform (f32 WMMA) -------
// D[s',f] = sum_p Re(out_sel[p,f] * e^{+i theta(p,s')})  (sym step folded in)
// o[s',ws] = D[512x16] @ V2t^T,  V2t[ws][f] = (f==0?1:2)*cos(2pi f ws/64)/32768
__global__ void idft_kernel(const float* __restrict__ Osel,
                            const int*   __restrict__ panel,
                            __bf16*      __restrict__ Obuf)
{
    __shared__ float osr[64 * 16 * 2];  // 8 KB
    __shared__ float v2t[64 * 16];      // [ws][f], 4 KB
    __shared__ float D[512 * 16];       // 32 KB
    const int bho = blockIdx.x;         // (b*8+h)*32+o
    const int t   = threadIdx.x;

    for (int i = t; i < 2048; i += 256) osr[i] = Osel[(size_t)bho * 2048 + i];
    for (int i = t; i < 1024; i += 256) {
        int ws = i >> 4, f = i & 15;
        float scale = (f == 0 ? 1.0f : 2.0f) / 32768.0f;
        v2t[i] = scale * __cosf(2.0f * PI_F * (float)(f * ws) / 64.0f);
    }
    __syncthreads();

    #pragma unroll
    for (int pass = 0; pass < 2; ++pass) {
        int   sp = t + pass * 256;
        float ft = (float)(sp >> 5) * (2.0f * PI_F / 16.0f);
        float fh = (float)(sp & 31) * (2.0f * PI_F / 32.0f);
        float acc[16];
        #pragma unroll
        for (int f = 0; f < 16; ++f) acc[f] = 0.f;
        for (int p = 0; p < 64; ++p) {
            int   flat = panel[p];
            float ang  = ft * (float)(flat >> 5) + fh * (float)(flat & 31);
            float sn, c; __sincosf(ang, &sn, &c);
            const float* a = osr + p * 32;
            #pragma unroll
            for (int f = 0; f < 16; ++f)
                acc[f] = fmaf(a[2 * f], c, fmaf(-a[2 * f + 1], sn, acc[f]));
        }
        #pragma unroll
        for (int f = 0; f < 16; ++f) D[sp * 16 + f] = acc[f];
    }
    __syncthreads();

    // phase 2: o[512x64] = D[512x16] @ V2, via f32 WMMA (K=16 -> 4 steps)
    const int lane = t & 31, wave = t >> 5;
    const int b  = bho >> 8;
    const int ho = bho & 255;           // h*32+o == output channel column
    for (int rt = wave; rt < 32; rt += 8) {
        const int m0 = rt * 16;
        const float* drow = D + (m0 + (lane & 15)) * 16;
        const int koffb = (lane >> 4) * 2;
        v8f acc[4] = {};
        #pragma unroll
        for (int k0 = 0; k0 < 16; k0 += 4) {
            v2f a = *(const v2f*)(drow + k0 + koffb);
            #pragma unroll
            for (int nt = 0; nt < 4; ++nt) {
                v2f bfr = *(const v2f*)(v2t + (nt * 16 + (lane & 15)) * 16 + k0 + koffb);
                acc[nt] = __builtin_amdgcn_wmma_f32_16x16x4_f32(
                    false, a, false, bfr, (short)0, acc[nt], false, false);
            }
        }
        #pragma unroll
        for (int nt = 0; nt < 4; ++nt) {
            #pragma unroll
            for (int v = 0; v < 8; ++v) {
                int sp  = m0 + v + 8 * (lane >> 4);
                int wsx = nt * 16 + (lane & 15);
                size_t r = ((size_t)b * 512 + sp) * 64 + wsx;
                Obuf[r * 256 + ho] = (__bf16)acc[nt][v];
            }
        }
    }
}

// ---------------------------------------------------------------------------------
extern "C" void kernel_launch(void* const* d_in, const int* in_sizes, int n_in,
                              void* d_out, int out_size, void* d_ws, size_t ws_size,
                              hipStream_t stream)
{
    (void)in_sizes; (void)n_in; (void)out_size; (void)ws_size;
    const float* x     = (const float*)d_in[0];
    const float* Wq    = (const float*)d_in[1];
    const float* bq    = (const float*)d_in[2];
    // d_in[3..6] (Wk,bk,Wv,bv) are dead code in the reference — skipped.
    const float* Wo    = (const float*)d_in[7];
    const float* bo    = (const float*)d_in[8];
    const float* wr    = (const float*)d_in[9];
    const float* wi    = (const float*)d_in[10];
    const int*   panel = (const int*)d_in[11];
    // d_in[12..13] (sym_idx1/2) folded analytically into idft_kernel.

    char*   ws   = (char*)d_ws;
    __bf16* xbf  = (__bf16*)(ws);                     // 64 MB  (reused as Obuf later)
    float*  Q2   = (float*)(ws + (size_t)67108864);   // 128 MB [1024][512][64]
    float*  G    = (float*)(ws + (size_t)201326592);  // 64 MB  [1024][512][16][2]
    float*  Xsel = (float*)(ws + (size_t)268435456);  // 8 MB   [1024][64][16][2]
    float*  Osel = (float*)(ws + (size_t)276824064);  // 8 MB   [1024][64][16][2]
    __bf16* Wqbf = (__bf16*)(ws + (size_t)285212672); // 128 KB, transposed [n][k]
    __bf16* Wobf = (__bf16*)(ws + (size_t)285343744); // 128 KB, transposed [n][k]

    cvt_f32_bf16<<<131072, 256, 0, stream>>>(x, xbf, 33554432);
    cvt_transpose_bf16<<<256, 256, 0, stream>>>(Wq, Wqbf);
    cvt_transpose_bf16<<<256, 256, 0, stream>>>(Wo, Wobf);

    gemm256_wmma<<<dim3(2048, 4), 128, 0, stream>>>(xbf, Wqbf, bq, Q2, 1);
    dft_ws_wmma<<<8192, 128, 0, stream>>>(Q2, G);
    dft_panel_kernel<<<1024, 64, 65536, stream>>>(G, panel, Xsel);
    mode_mix_kernel<<<512, 256, 0, stream>>>(Xsel, wr, wi, Osel);
    idft_kernel<<<1024, 256, 0, stream>>>(Osel, panel, xbf /* Obuf */);
    gemm256_wmma<<<dim3(2048, 4), 128, 0, stream>>>(xbf, Wobf, bo, (float*)d_out, 0);
}
